// GraphSAGE_82497731821612
// MI455X (gfx1250) — compile-verified, matrix operand order checked by reading, not run on hardware
//
#include <hip/hip_runtime.h>
#include <stdint.h>
#include <stddef.h>

#define NND 100000

typedef __bf16 bf16_t;
typedef bf16_t  v16bf __attribute__((ext_vector_type(16)));
typedef float   v8f   __attribute__((ext_vector_type(8)));
typedef uint32_t v8u  __attribute__((ext_vector_type(8)));

__device__ __forceinline__ uint32_t bf16pack2(float lo, float hi) {
    uint32_t a = __builtin_bit_cast(uint32_t, lo);
    uint32_t b = __builtin_bit_cast(uint32_t, hi);
    a += 0x7fffu + ((a >> 16) & 1u);   // round-to-nearest-even
    b += 0x7fffu + ((b >> 16) & 1u);
    return (a >> 16) | (b & 0xffff0000u);
}

// ---------------------------------------------------------------- utilities
__global__ void gs_zero_kernel(float* __restrict__ p, int n) {
    int i = blockIdx.x * blockDim.x + threadIdx.x;
    if (i < n) p[i] = 0.0f;
}

__global__ void gs_degree_kernel(const long long* __restrict__ dst, int E,
                                 float* __restrict__ counts) {
    int e = blockIdx.x * blockDim.x + threadIdx.x;
    if (e < E) atomicAdd(&counts[(int)dst[e]], 1.0f);
}

__global__ void gs_invcount_kernel(float* __restrict__ c, int n) {
    int i = blockIdx.x * blockDim.x + threadIdx.x;
    if (i < n) c[i] = 1.0f / fmaxf(c[i], 1.0f);
}

// one thread per (edge, quad-of-4-features): float4 gather + 4 f32 atomics
__global__ void gs_scatter_kernel(const float* __restrict__ x,
                                  const long long* __restrict__ src,
                                  const long long* __restrict__ dst,
                                  int E, float* __restrict__ aggr) {
    int idx = blockIdx.x * blockDim.x + threadIdx.x;
    int e = idx >> 5;
    int q = idx & 31;
    if (e >= E) return;
    int s = (int)src[e];
    int d = (int)dst[e];
    const float4 v = *(const float4*)(x + (size_t)s * 128 + q * 4);
    float* base = aggr + (size_t)d * 128 + q * 4;
    atomicAdd(base + 0, v.x);
    atomicAdd(base + 1, v.y);
    atomicAdd(base + 2, v.z);
    atomicAdd(base + 3, v.w);
}

// ------------------------------------------------- pack weights into B-frag
// layout: dword index (((nt*4+kk)*32 + lane)*8 + v), pair (k0,k0+1) for col n
__global__ void gs_packB_kernel(const float* __restrict__ W, int ldw,
                                int ncols_real, int ntiles,
                                uint32_t* __restrict__ Bp) {
    int idx = blockIdx.x * blockDim.x + threadIdx.x;
    int total = ntiles * 4 * 32 * 8;
    if (idx >= total) return;
    int v    = idx & 7;
    int lane = (idx >> 3) & 31;
    int kk   = (idx >> 8) & 3;
    int nt   = idx >> 10;
    int n  = lane & 15;
    int hf = lane >> 4;
    int k0 = kk * 32 + ((v >= 4) ? 16 : 0) + hf * 8 + (v & 3) * 2;
    int col = nt * 16 + n;
    float w0 = 0.0f, w1 = 0.0f;
    if (col < ncols_real) {
        w0 = W[(size_t)k0 * ldw + col];
        w1 = W[(size_t)(k0 + 1) * ldw + col];
    }
    Bp[idx] = bf16pack2(w0, w1);
}

// ---------------------------------------------------------------- WMMA GEMM
// out[r, :] = act( (A1[r,:]*invc[r]) @ B1  (+ A2[r,:] @ B2)  + bias )
// K = 128 always. 128 threads = 4 waves, each wave owns 16 rows.
template <int NTILES, bool DUAL, bool RELU>
__global__ __launch_bounds__(128) void gs_gemm_kernel(
    const float* __restrict__ A1, const float* __restrict__ invc,
    const uint32_t* __restrict__ Bp1,
    const float* __restrict__ A2, const uint32_t* __restrict__ Bp2,
    const float* __restrict__ bias, int ncols_real,
    float* __restrict__ out, int ldout, int nrows) {
    const int lane = threadIdx.x & 31;
    const int wave = threadIdx.x >> 5;
    const int m  = lane & 15;
    const int hf = lane >> 4;
    const int row_base = blockIdx.x * 64 + wave * 16;
    int rowA = row_base + m;
    if (rowA >= nrows) rowA = nrows - 1;          // clamp: EXEC stays all-1s
    const float scale = invc ? invc[rowA] : 1.0f;

    // per-lane bias values for every column tile, hoisted out of the MMA loop
    float bv[NTILES];
    #pragma unroll
    for (int nt = 0; nt < NTILES; ++nt) {
        int col = nt * 16 + m;
        bv[nt] = (col < ncols_real) ? bias[col] : 0.0f;
    }

    v16bf a1[4];
    v16bf a2[4] = {};
    #pragma unroll
    for (int kk = 0; kk < 4; ++kk) {
        const float* r = A1 + (size_t)rowA * 128 + kk * 32 + hf * 8;
        float4 f0 = *(const float4*)(r);
        float4 f1 = *(const float4*)(r + 4);
        float4 f2 = *(const float4*)(r + 16);
        float4 f3 = *(const float4*)(r + 20);
        v8u u;
        u[0] = bf16pack2(f0.x * scale, f0.y * scale);
        u[1] = bf16pack2(f0.z * scale, f0.w * scale);
        u[2] = bf16pack2(f1.x * scale, f1.y * scale);
        u[3] = bf16pack2(f1.z * scale, f1.w * scale);
        u[4] = bf16pack2(f2.x * scale, f2.y * scale);
        u[5] = bf16pack2(f2.z * scale, f2.w * scale);
        u[6] = bf16pack2(f3.x * scale, f3.y * scale);
        u[7] = bf16pack2(f3.z * scale, f3.w * scale);
        a1[kk] = __builtin_bit_cast(v16bf, u);
    }
    if constexpr (DUAL) {
        #pragma unroll
        for (int kk = 0; kk < 4; ++kk) {
            const float* r = A2 + (size_t)rowA * 128 + kk * 32 + hf * 8;
            float4 f0 = *(const float4*)(r);
            float4 f1 = *(const float4*)(r + 4);
            float4 f2 = *(const float4*)(r + 16);
            float4 f3 = *(const float4*)(r + 20);
            v8u u;
            u[0] = bf16pack2(f0.x, f0.y);
            u[1] = bf16pack2(f0.z, f0.w);
            u[2] = bf16pack2(f1.x, f1.y);
            u[3] = bf16pack2(f1.z, f1.w);
            u[4] = bf16pack2(f2.x, f2.y);
            u[5] = bf16pack2(f2.z, f2.w);
            u[6] = bf16pack2(f3.x, f3.y);
            u[7] = bf16pack2(f3.z, f3.w);
            a2[kk] = __builtin_bit_cast(v16bf, u);
        }
    }

    #pragma unroll
    for (int nt = 0; nt < NTILES; ++nt) {
        // preload the whole tile's B fragments first -> clause of b128 loads,
        // then an uninterrupted chain of WMMAs; unrolled tiles let the
        // scheduler pull the next tile's loads under this tile's math.
        v16bf b1f[4];
        #pragma unroll
        for (int kk = 0; kk < 4; ++kk) {
            v8u bu = *(const v8u*)(Bp1 + ((((nt * 4 + kk) * 32) + lane) << 3));
            b1f[kk] = __builtin_bit_cast(v16bf, bu);
        }
        v16bf b2f[4] = {};
        if constexpr (DUAL) {
            #pragma unroll
            for (int kk = 0; kk < 4; ++kk) {
                v8u bu = *(const v8u*)(Bp2 + ((((nt * 4 + kk) * 32) + lane) << 3));
                b2f[kk] = __builtin_bit_cast(v16bf, bu);
            }
        }
        v8f acc = {};
        #pragma unroll
        for (int kk = 0; kk < 4; ++kk)
            acc = __builtin_amdgcn_wmma_f32_16x16x32_bf16(
                false, a1[kk], false, b1f[kk], (short)0, acc, false, false);
        if constexpr (DUAL) {
            #pragma unroll
            for (int kk = 0; kk < 4; ++kk)
                acc = __builtin_amdgcn_wmma_f32_16x16x32_bf16(
                    false, a2[kk], false, b2f[kk], (short)0, acc, false, false);
        }
        const int col = nt * 16 + m;
        #pragma unroll
        for (int r = 0; r < 8; ++r) {
            int orow = row_base + r + 8 * hf;   // D layout: vgpr r -> M = r + 8*half
            float vv = acc[r] + bv[nt];
            if constexpr (RELU) vv = fmaxf(vv, 0.0f);
            if (orow < nrows && col < ncols_real)
                out[(size_t)orow * ldout + col] = vv;
        }
    }
}

// ------------------------------------------------------------- log-softmax
__global__ void gs_logsoftmax_kernel(const float* __restrict__ z,
                                     float* __restrict__ out, int nrows) {
    int i = blockIdx.x * blockDim.x + threadIdx.x;
    if (i >= nrows) return;
    const float* r = z + (size_t)i * 40;
    float mx = -3.4e38f;
    #pragma unroll
    for (int c = 0; c < 40; ++c) mx = fmaxf(mx, r[c]);
    float s = 0.0f;
    #pragma unroll
    for (int c = 0; c < 40; ++c) s += expf(r[c] - mx);
    float ls = logf(s);
    float* o = out + (size_t)i * 40;
    #pragma unroll
    for (int c = 0; c < 40; ++c) o[c] = (r[c] - mx) - ls;
}

// ------------------------------------------------------------------ launch
extern "C" void kernel_launch(void* const* d_in, const int* in_sizes, int n_in,
                              void* d_out, int out_size, void* d_ws, size_t ws_size,
                              hipStream_t stream) {
    (void)n_in; (void)out_size; (void)ws_size;
    const int N = NND;
    const int E = in_sizes[1] / 2;

    const float*     x    = (const float*)d_in[0];
    const long long* ei   = (const long long*)d_in[1];
    const long long* esrc = ei;
    const long long* edst = ei + E;
    const float* W1l = (const float*)d_in[2];
    const float* b1  = (const float*)d_in[3];
    const float* W1r = (const float*)d_in[4];
    const float* W2l = (const float*)d_in[5];
    const float* b2  = (const float*)d_in[6];
    const float* W2r = (const float*)d_in[7];
    const float* W3l = (const float*)d_in[8];
    const float* b3  = (const float*)d_in[9];
    const float* W3r = (const float*)d_in[10];
    const float* Wf1 = (const float*)d_in[11];
    const float* bf1 = (const float*)d_in[12];
    const float* Wf2 = (const float*)d_in[13];
    const float* bf2 = (const float*)d_in[14];

    // workspace layout
    float* invc = (float*)d_ws;                      // N
    float* aggr = invc + N;                          // N*128
    float* h1   = aggr + (size_t)N * 128;            // N*128
    float* h2   = h1   + (size_t)N * 128;            // N*128
    uint32_t* BpL = (uint32_t*)(h2 + (size_t)N * 128);   // 8192 dwords (32B aligned)
    uint32_t* BpR = BpL + 8192;                      // 8192 dwords

    float* out = (float*)d_out;                      // N*40 log-softmax
    float* h3  = out + (size_t)N * 40;               // N*128 (second output)

    const int TB = 256;
    const int NF = N * 128;
    const int gemm_blocks = (N + 63) / 64;

    // degree counts -> inverse
    gs_zero_kernel<<<(N + TB - 1) / TB, TB, 0, stream>>>(invc, N);
    gs_degree_kernel<<<(E + TB - 1) / TB, TB, 0, stream>>>(edst, E, invc);
    gs_invcount_kernel<<<(N + TB - 1) / TB, TB, 0, stream>>>(invc, N);

    auto sage_pre = [&](const float* xin, const float* Wl, const float* Wr) {
        gs_zero_kernel<<<(NF + TB - 1) / TB, TB, 0, stream>>>(aggr, NF);
        gs_scatter_kernel<<<(E * 32 + TB - 1) / TB, TB, 0, stream>>>(
            xin, esrc, edst, E, aggr);
        gs_packB_kernel<<<(8192 + TB - 1) / TB, TB, 0, stream>>>(Wl, 128, 128, 8, BpL);
        gs_packB_kernel<<<(8192 + TB - 1) / TB, TB, 0, stream>>>(Wr, 128, 128, 8, BpR);
    };

    // layer 1: relu
    sage_pre(x, W1l, W1r);
    gs_gemm_kernel<8, true, true><<<gemm_blocks, 128, 0, stream>>>(
        aggr, invc, BpL, x, BpR, b1, 128, h1, 128, N);
    // layer 2: relu
    sage_pre(h1, W2l, W2r);
    gs_gemm_kernel<8, true, true><<<gemm_blocks, 128, 0, stream>>>(
        aggr, invc, BpL, h1, BpR, b2, 128, h2, 128, N);
    // layer 3: linear, h3 written straight into d_out (second output)
    sage_pre(h2, W3l, W3r);
    gs_gemm_kernel<8, true, false><<<gemm_blocks, 128, 0, stream>>>(
        aggr, invc, BpL, h2, BpR, b3, 128, h3, 128, N);

    // MLP head: z1 = relu(h3 @ Wf1 + bf1)  (reuse h1)
    gs_packB_kernel<<<(8192 + TB - 1) / TB, TB, 0, stream>>>(Wf1, 128, 128, 8, BpL);
    gs_gemm_kernel<8, false, true><<<gemm_blocks, 128, 0, stream>>>(
        h3, nullptr, BpL, nullptr, nullptr, bf1, 128, h1, 128, N);

    // z2 = z1 @ Wf2 + bf2  (40 cols, 3 padded tiles; reuse aggr)
    gs_packB_kernel<<<(3072 + TB - 1) / TB, TB, 0, stream>>>(Wf2, 40, 40, 3, BpL);
    gs_gemm_kernel<3, false, false><<<gemm_blocks, 128, 0, stream>>>(
        h1, nullptr, BpL, nullptr, nullptr, bf2, 40, aggr, 40, N);

    gs_logsoftmax_kernel<<<(N + TB - 1) / TB, TB, 0, stream>>>(aggr, out, N);
}
